// BinaryConv_72155450573462
// MI455X (gfx1250) — matrix operand order repaired and checked by reading, not compiled, not run on hardware
//
#include <hip/hip_runtime.h>
#include <math.h>

typedef __bf16 bf16;
typedef __attribute__((ext_vector_type(16))) __bf16 v16bf;
typedef __attribute__((ext_vector_type(8)))  __bf16 v8bf;
typedef __attribute__((ext_vector_type(8)))  float  v8f;

__device__ __forceinline__ float signf3(float w) {
    return (w > 0.f) ? 1.f : ((w < 0.f) ? -1.f : 0.f);
}

// ---------------- prep kernels ----------------

__global__ void cvt_f32_to_bf16(const float* __restrict__ in, bf16* __restrict__ out, int n) {
    int i = blockIdx.x * blockDim.x + threadIdx.x;
    if (i < n) out[i] = (bf16)in[i];
}

// Binarize + repack weights into out[kt][n][32], slot linear in VIRTUAL k.
// Virtual k ordering: v = rs*CinP + ci  (ci fastest). Physical weight index:
// n*K + ci*RSp + rs. CinP=K, RSp=1 gives the identity permutation.
// B fragment (32x16 KxN): lane l -> column n=l%16, half l/16 covers
// v = kt*32 + half*16 + 0..15 => 32 contiguous bytes per lane.
__global__ void prep_weight_sw(const float* __restrict__ w, bf16* __restrict__ out,
                               int N, int K, int CinP, int RSp, int total) {
    int i = blockIdx.x * blockDim.x + threadIdx.x;
    if (i >= total) return;
    int slot = i & 31;
    int tn   = i >> 5;
    int n    = tn % N;
    int kt   = tn / N;
    int v    = kt * 32 + slot;
    float val = 0.f;
    if (v < K) {
        int rs = v / CinP;
        int ci = v - rs * CinP;
        val = signf3(w[(size_t)n * K + ci * RSp + rs]);
    }
    out[i] = (bf16)val;
}

// Fold conv bias + batchnorm into y = conv*scale + shift
__global__ void prep_bn(const float* __restrict__ bias, const float* __restrict__ g,
                        const float* __restrict__ be, const float* __restrict__ m,
                        const float* __restrict__ v,
                        float* __restrict__ scale, float* __restrict__ shift, int C) {
    int i = blockIdx.x * blockDim.x + threadIdx.x;
    if (i < C) {
        float inv = g[i] * rsqrtf(v[i] + 1e-5f);
        scale[i] = inv;
        shift[i] = bias[i] * inv + be[i] - m[i] * inv;
    }
}

__global__ void prep_identity(const float* __restrict__ bias,
                              float* __restrict__ scale, float* __restrict__ shift, int C) {
    int i = blockIdx.x * blockDim.x + threadIdx.x;
    if (i < C) { scale[i] = 1.f; shift[i] = bias[i]; }
}

// ---------------- implicit-GEMM conv / FC via bf16 WMMA ----------------
// M = B*OH*OW, N = Cout, K = Cin*RS.
// Block: 256 threads = 8 waves tiled 4(M) x 2(N); each wave: one 16-row M tile,
// TWO 16-col N tiles => block tile 64x64, 2 WMMAs/wave/step.
// A tile double-buffered in LDS in exact WMMA A-fragment order.
// Fast path (RS==9, !GUARD, Cin%32==0): virtual-K order v=rs*Cin+ci, so each
// k-step has uniform (r,s), and thread elements are ci,ci+1,..,ci+7 (stride HW).
template <int RS, bool GUARD>
__global__ __launch_bounds__(256)
void conv_gemm_wmma(const bf16* __restrict__ xin,
                    const bf16* __restrict__ wsw,     // [ceil(K/32)][Cout][32]
                    const float* __restrict__ scale,
                    const float* __restrict__ shift,
                    bf16* __restrict__ out,           // NCHW bf16
                    int Cin, int H, int W, int Cout,
                    int OH, int OW, int K, int relu)
{
    const int OHW = OH * OW;
    const int t = threadIdx.x;
    const int blockM = blockIdx.x * 64;
    const int blockN = blockIdx.y * 64;

    __shared__ __align__(32) bf16 As[2][4 * 32 * 16];   // double-buffered

    // ---- cooperative A-gather: thread t owns (mt, lane, 8 consecutive k) ----
    const int a_mt = t >> 6;
    const int a_ln = (t >> 1) & 31;
    const int a_e0 = (t & 1) * 8;                     // 0 or 8
    const int a_hi = a_ln >> 4;
    const int a_kbase = a_hi * 16 + a_e0;             // 8 consecutive virtual k
    const int a_m   = blockM + a_mt * 16 + (a_ln & 15);
    const int a_b   = a_m / OHW;
    const int a_rem = a_m % OHW;
    const int a_oh  = a_rem / OW;
    const int a_ow  = a_rem % OW;
    const bf16* xbase = xin + ((size_t)a_b * Cin * H + a_oh) * W + a_ow;

    // ---- wave tiling ----
    const int wv  = t >> 5;
    const int ln  = t & 31;
    const int mtw = wv >> 1;
    const int ntw = wv & 1;
    const int n0  = blockN + ntw * 32 + (ln & 15);
    const int n1  = n0 + 16;
    const int bhi = ln >> 4;

    const int Ksteps = (K + 31) >> 5;

    v8f acc0 = {}, acc1 = {};

    const size_t HW = (size_t)H * W;
    // fast-path: uniform virtual-k walk state (pure scalar)
    int u_cb = 0;            // ci base of current k-tile
    int u_s  = 0;
    int u_rowoff = 0;        // r*W + s
    // loop-invariant lane offset (elements)
    const int lanebase = (int)(((size_t)a_b * Cin * H + a_oh) * W + a_ow
                               + (size_t)a_kbase * HW);

    auto do_gather = [&](int ks) -> v8bf {
        (void)ks;
        v8bf sv;
        if constexpr (RS == 1) {
            sv = *reinterpret_cast<const v8bf*>(xbase + (ks << 5) + a_kbase);
        } else if constexpr (!GUARD) {
            const bf16* pu = xin + (size_t)u_cb * HW + u_rowoff;   // uniform ptr
            #pragma unroll
            for (int j = 0; j < 8; ++j)
                sv[j] = pu[(size_t)j * HW + lanebase];             // saddr + lane off
            // advance uniform state (SALU)
            u_cb += 32;
            if (u_cb == Cin) {
                u_cb = 0;
                ++u_s; ++u_rowoff;
                if (u_s == 3) { u_s = 0; u_rowoff += W - 3; }
            }
        } else {
            const int k0 = (ks << 5) + a_kbase;
            int ci = k0 / 9;
            int rs = k0 - ci * 9;
            int r  = rs / 3;
            int s  = rs - r * 3;
            int off = (ci * H + r) * W + s;
            #pragma unroll
            for (int j = 0; j < 8; ++j) {
                const bool ok = (k0 + j) < K;
                bf16 vld = xbase[ok ? off : 0];
                sv[j] = ok ? vld : (bf16)0.f;
                const bool wS = (s == 2);
                const bool wR = wS && (r == 2);
                off += 1 + (wS ? (W - 3) : 0) + (wR ? (H - 3) * W : 0);
                s = wS ? 0 : s + 1;
                r = wS ? (wR ? 0 : r + 1) : r;
            }
        }
        return sv;
    };

    v8bf sv = do_gather(0);
    for (int ks = 0; ks < Ksteps; ++ks) {
        bf16* asbuf = As[ks & 1];
        *reinterpret_cast<v8bf*>(&asbuf[t * 8]) = sv;          // ds_store_b128
        if (ks + 1 < Ksteps) sv = do_gather(ks + 1);           // prefetch next
        const bf16* bp = wsw + (size_t)ks * Cout * 32;
        v16bf bf0 = *reinterpret_cast<const v16bf*>(bp + (size_t)n0 * 32 + bhi * 16);
        v16bf bf1 = *reinterpret_cast<const v16bf*>(bp + (size_t)n1 * 32 + bhi * 16);
        __syncthreads();                                       // single barrier / step
        v16bf afrag = *reinterpret_cast<const v16bf*>(&asbuf[(mtw * 32 + ln) * 16]);
        acc0 = __builtin_amdgcn_wmma_f32_16x16x32_bf16(
            false, afrag, false, bf0, (short)0, acc0, false, false);
        acc1 = __builtin_amdgcn_wmma_f32_16x16x32_bf16(
            false, afrag, false, bf1, (short)0, acc1, false, false);
    }

    // ---- epilogue: scale/shift(+ReLU), store bf16 NCHW ----
    const float sc0 = scale[n0], sh0 = shift[n0];
    const float sc1 = scale[n1], sh1 = shift[n1];
    const int mbase = blockM + mtw * 16 + bhi * 8;             // C layout: VGPR r -> row
    int b   = mbase / OHW;
    int rem = mbase - b * OHW;
    size_t bbase = (size_t)b * Cout * OHW;
    const size_t chstep = (size_t)Cout * OHW;
    #pragma unroll
    for (int r = 0; r < 8; ++r) {
        float v0 = acc0[r] * sc0 + sh0;
        float v1 = acc1[r] * sc1 + sh1;
        if (relu) { v0 = fmaxf(v0, 0.f); v1 = fmaxf(v1, 0.f); }
        out[bbase + (size_t)n0 * OHW + rem] = (bf16)v0;
        out[bbase + (size_t)n1 * OHW + rem] = (bf16)v1;
        ++rem;
        const bool wrap = (rem == OHW);
        rem = wrap ? 0 : rem;
        bbase += wrap ? chstep : 0;
    }
}

// ---------------- 2x2 max pool (bf16) ----------------
__global__ void pool2x2(const bf16* __restrict__ in, bf16* __restrict__ out,
                        int C, int H, int W, int total) {
    int i = blockIdx.x * blockDim.x + threadIdx.x;
    if (i >= total) return;
    int OW = W >> 1, OH = H >> 1;
    int ow = i % OW; int tt = i / OW;
    int oh = tt % OH; tt /= OH;
    int c = tt % C;  int b = tt / C;
    const bf16* p = in + ((size_t)(b * C + c) * H + oh * 2) * W + ow * 2;
    float v0 = (float)p[0], v1 = (float)p[1], v2 = (float)p[W], v3 = (float)p[W + 1];
    out[i] = (bf16)fmaxf(fmaxf(v0, v1), fmaxf(v2, v3));
}

// ---------------- final FC (N=10) + softmax ----------------
__global__ __launch_bounds__(256)
void fc3_softmax(const bf16* __restrict__ x,   // [512][1024]
                 const float* __restrict__ w,  // [10][1024]
                 const float* __restrict__ bias,
                 float* __restrict__ out) {    // [512][10]
    const int b = blockIdx.x;
    const bf16* xr = x + (size_t)b * 1024;
    float s[10];
    #pragma unroll
    for (int nn = 0; nn < 10; ++nn) s[nn] = 0.f;

    for (int k = threadIdx.x; k < 1024; k += 256) {
        float xv = (float)xr[k];
        #pragma unroll
        for (int nn = 0; nn < 10; ++nn) {
            s[nn] += xv * signf3(w[nn * 1024 + k]);
        }
    }
    #pragma unroll
    for (int off = 16; off > 0; off >>= 1) {
        #pragma unroll
        for (int nn = 0; nn < 10; ++nn) s[nn] += __shfl_down(s[nn], off, 32);
    }
    __shared__ float red[8][10];
    int wv = threadIdx.x >> 5, ln = threadIdx.x & 31;
    if (ln == 0) {
        #pragma unroll
        for (int nn = 0; nn < 10; ++nn) red[wv][nn] = s[nn];
    }
    __syncthreads();
    if (threadIdx.x == 0) {
        float tot[10];
        #pragma unroll
        for (int nn = 0; nn < 10; ++nn) {
            float a = bias[nn];
            for (int ww = 0; ww < 8; ++ww) a += red[ww][nn];
            tot[nn] = a;
        }
        float mx = tot[0];
        #pragma unroll
        for (int nn = 1; nn < 10; ++nn) mx = fmaxf(mx, tot[nn]);
        float se = 0.f;
        #pragma unroll
        for (int nn = 0; nn < 10; ++nn) { tot[nn] = expf(tot[nn] - mx); se += tot[nn]; }
        float inv = 1.f / se;
        #pragma unroll
        for (int nn = 0; nn < 10; ++nn) out[(size_t)b * 10 + nn] = tot[nn] * inv;
    }
}

// ---------------- host orchestration ----------------

extern "C" void kernel_launch(void* const* d_in, const int* in_sizes, int n_in,
                              void* d_out, int out_size, void* d_ws, size_t ws_size,
                              hipStream_t stream) {
    (void)in_sizes; (void)n_in; (void)out_size; (void)ws_size;

    const float* x = (const float*)d_in[0];
    const float *w[6], *cb[6], *cg[6], *cbe[6], *cm[6], *cv[6];
    for (int i = 0; i < 6; ++i) {
        w[i]   = (const float*)d_in[1 + 6 * i + 0];
        cb[i]  = (const float*)d_in[1 + 6 * i + 1];
        cg[i]  = (const float*)d_in[1 + 6 * i + 2];
        cbe[i] = (const float*)d_in[1 + 6 * i + 3];
        cm[i]  = (const float*)d_in[1 + 6 * i + 4];
        cv[i]  = (const float*)d_in[1 + 6 * i + 5];
    }
    const float* fw1 = (const float*)d_in[37];
    const float* fb1 = (const float*)d_in[38];
    const float* fw2 = (const float*)d_in[39];
    const float* fb2 = (const float*)d_in[40];
    const float* fw3 = (const float*)d_in[41];
    const float* fb3 = (const float*)d_in[42];
    float* out = (float*)d_out;

    char* base = (char*)d_ws;
    size_t off = 0;
    auto alloc = [&](size_t bytes) -> void* {
        void* p = base + off;
        off = (off + bytes + 255) & ~(size_t)255;
        return p;
    };

    const int Bn = 512;
    const int Cin[6]  = {3, 128, 128, 256, 256, 512};
    const int Hh[6]   = {32, 30, 14, 12, 5, 3};
    const int Cout[6] = {128, 128, 256, 256, 512, 512};
    const int OHa[6]  = {30, 28, 12, 10, 3, 1};
    int Kc[6];
    for (int i = 0; i < 6; ++i) Kc[i] = Cin[i] * 9;

    bf16* xbf  = (bf16*)alloc((size_t)Bn * 3 * 32 * 32 * sizeof(bf16));
    bf16* bufA = (bf16*)alloc((size_t)Bn * 128 * 30 * 30 * sizeof(bf16));
    bf16* bufB = (bf16*)alloc((size_t)Bn * 128 * 28 * 28 * sizeof(bf16));

    bf16*  wsw[8];
    float* lsc[8];
    float* lsh[8];
    int    wtot[8];
    for (int i = 0; i < 6; ++i) {
        int kt = (Kc[i] + 31) / 32;
        wtot[i] = kt * Cout[i] * 32;
        wsw[i] = (bf16*)alloc((size_t)wtot[i] * sizeof(bf16));
        lsc[i] = (float*)alloc((size_t)Cout[i] * sizeof(float));
        lsh[i] = (float*)alloc((size_t)Cout[i] * sizeof(float));
    }
    wtot[6] = (512 / 32) * 1024 * 32;
    wtot[7] = (1024 / 32) * 1024 * 32;
    wsw[6] = (bf16*)alloc((size_t)wtot[6] * sizeof(bf16));
    wsw[7] = (bf16*)alloc((size_t)wtot[7] * sizeof(bf16));
    lsc[6] = (float*)alloc(1024 * sizeof(float));
    lsh[6] = (float*)alloc(1024 * sizeof(float));
    lsc[7] = (float*)alloc(1024 * sizeof(float));
    lsh[7] = (float*)alloc(1024 * sizeof(float));

    // ---- prep ----
    {
        int n = Bn * 3 * 32 * 32;
        cvt_f32_to_bf16<<<(n + 255) / 256, 256, 0, stream>>>(x, xbf, n);
    }
    // conv1: identity permutation (CinP=K, RSp=1) -- GUARD kernel decodes physical k
    prep_weight_sw<<<(wtot[0] + 255) / 256, 256, 0, stream>>>(w[0], wsw[0], Cout[0], Kc[0],
                                                              Kc[0], 1, wtot[0]);
    prep_bn<<<(Cout[0] + 255) / 256, 256, 0, stream>>>(cb[0], cg[0], cbe[0], cm[0], cv[0],
                                                       lsc[0], lsh[0], Cout[0]);
    // conv2..6: virtual order v = rs*Cin + ci (ci fastest)
    for (int i = 1; i < 6; ++i) {
        prep_weight_sw<<<(wtot[i] + 255) / 256, 256, 0, stream>>>(w[i], wsw[i], Cout[i], Kc[i],
                                                                  Cin[i], 9, wtot[i]);
        prep_bn<<<(Cout[i] + 255) / 256, 256, 0, stream>>>(cb[i], cg[i], cbe[i], cm[i], cv[i],
                                                           lsc[i], lsh[i], Cout[i]);
    }
    // fc1/fc2: identity permutation
    prep_weight_sw<<<(wtot[6] + 255) / 256, 256, 0, stream>>>(fw1, wsw[6], 1024, 512,
                                                              512, 1, wtot[6]);
    prep_weight_sw<<<(wtot[7] + 255) / 256, 256, 0, stream>>>(fw2, wsw[7], 1024, 1024,
                                                              1024, 1, wtot[7]);
    prep_identity<<<4, 256, 0, stream>>>(fb1, lsc[6], lsh[6], 1024);
    prep_identity<<<4, 256, 0, stream>>>(fb2, lsc[7], lsh[7], 1024);

    // ---- helpers ----
    auto convN = [&](const bf16* in, int li, bf16* o) {   // Cin%32==0 layers
        int M = Bn * OHa[li] * OHa[li];
        dim3 grid(M / 64, Cout[li] / 64);
        conv_gemm_wmma<9, false><<<grid, 256, 0, stream>>>(
            in, wsw[li], lsc[li], lsh[li], o,
            Cin[li], Hh[li], Hh[li], Cout[li], OHa[li], OHa[li], Kc[li], 1);
    };
    auto fc = [&](const bf16* in, int wi, int Kk, int Nn, bf16* o) {
        dim3 grid(Bn / 64, Nn / 64);
        conv_gemm_wmma<1, false><<<grid, 256, 0, stream>>>(
            in, wsw[wi], lsc[wi], lsh[wi], o,
            Kk, 1, 1, Nn, 1, 1, Kk, 1);
    };

    // ---- network ----
    {   // conv1: K=27, guarded general path
        int M = Bn * 30 * 30;
        dim3 grid(M / 64, 128 / 64);
        conv_gemm_wmma<9, true><<<grid, 256, 0, stream>>>(
            xbf, wsw[0], lsc[0], lsh[0], bufA,
            3, 32, 32, 128, 30, 30, 27, 1);
    }                                                          // [512,128,30,30]
    convN(bufA, 1, bufB);                                      // [512,128,28,28]
    {
        int total = Bn * 128 * 14 * 14;
        pool2x2<<<(total + 255) / 256, 256, 0, stream>>>(bufB, bufA, 128, 28, 28, total);
    }                                                          // [512,128,14,14]
    convN(bufA, 2, bufB);                                      // [512,256,12,12]
    convN(bufB, 3, bufA);                                      // [512,256,10,10]
    {
        int total = Bn * 256 * 5 * 5;
        pool2x2<<<(total + 255) / 256, 256, 0, stream>>>(bufA, bufB, 256, 10, 10, total);
    }                                                          // [512,256,5,5]
    convN(bufB, 4, bufA);                                      // [512,512,3,3]
    convN(bufA, 5, bufB);                                      // [512,512,1,1]
    fc(bufB, 6, 512, 1024, bufA);                              // [512,1024]
    fc(bufA, 7, 1024, 1024, bufB);                             // [512,1024]
    fc3_softmax<<<Bn, 256, 0, stream>>>(bufB, fw3, fb3, out);  // [512,10]
}
